// OptimizedWindowAttention2D_27247272526245
// MI455X (gfx1250) — compile-verified, matrix operand order checked by reading, not run on hardware
//
#include <hip/hip_runtime.h>

// ---------------------------------------------------------------------------
// FULLY-FUSED window attention (Swin-style) for gfx1250 / MI455X.
// One workgroup per 8x8 window: QKV (WMMA) -> 4x head attention (WMMA +
// register softmax) -> output projection (WMMA) -> [B,C,H,W] store.
// No global scratch: eliminates ~540MB of HBM round trips vs. 3-kernel
// version -> ~268MB total traffic ~= 11.5us floor at 23.3 TB/s.
// LDS: 109072 B with phase-based region reuse (ISA allows 320KB/workgroup).
// ---------------------------------------------------------------------------

typedef __attribute__((ext_vector_type(16))) _Float16 v16h;
typedef __attribute__((ext_vector_type(8)))  _Float16 v8h;
typedef __attribute__((ext_vector_type(4)))  _Float16 v4h;
typedef __attribute__((ext_vector_type(8)))  float    v8f;
typedef __attribute__((ext_vector_type(4)))  float    v4f;
typedef __attribute__((ext_vector_type(4)))  int      v4i;

#define NWIN  4096   // (256/8)*(256/8)*4
#define HEADS 4
#define HD    32
#define NTOK  64
#define CCH   128

// 16-bit WMMA operand fragment (ISA 7.12.2): per-lane row = row0 + lane%16,
// K-pairs for lane-half h are two contiguous 8-half runs at k0+h*8 and
// k0+16+h*8 -> two 16-byte LDS loads. Requires ld % 8 == 0.
// Works for A (row = M) and for B stored as [N][K] (row = N).
__device__ __forceinline__ v16h frag_ld(const _Float16* buf, int row0, int k0,
                                        int ld, int lane) {
  const _Float16* r = buf + (row0 + (lane & 15)) * ld + k0 + (lane >> 4) * 8;
  union { v16h f; v8h h[2]; } u;
  u.h[0] = *(const v8h*)r;
  u.h[1] = *(const v8h*)(r + 16);
  return u.f;
}

// LDS region offsets (bytes); all 16B-aligned.
#define SM_XS    0       // phase0: x window [p][c] f16 ld136 (17408B)
                         // phase1: O output  [p][c] f16 ld136
                         // phase2: proj A operand
#define SM_WSL   17408   // phase0: w_qkv chunk [c3][c] f16 ld136 (34816B)
                         // phase1: Sf f32 [m][n] ld68 (17408B) + Ps @34816
                         // phase2: w_proj [co][ci] f16 ld136 (34816B)
#define SM_PS    34816   // phase1: Ps f16 [m][n] ld72 (9216B)
#define SM_QS    52224   // q f16 [p][c] ld136 (17408B); phase2: os f32 (part)
#define SM_KS    69632   // k f16 [p][c] ld136 (17408B); phase2: os f32 (part)
#define SM_VT    87040   // v^T f16 [c][m] ld72 (18432B)
#define SM_BT    105472  // bias_table copy, 900 floats (3600B)
#define SM_TOTAL 109072

__global__ __launch_bounds__(256) void fused_window_attn_kernel(
    const float* __restrict__ x, const float* __restrict__ w_qkv,
    const float* __restrict__ bias_table, const int* __restrict__ rel_index,
    const float* __restrict__ w_proj, const float* __restrict__ b_proj,
    float* __restrict__ out) {
  __shared__ __align__(16) unsigned char smem[SM_TOTAL];
  _Float16* xs  = (_Float16*)(smem + SM_XS);   // also O / proj-A
  _Float16* wsl = (_Float16*)(smem + SM_WSL);  // also wp
  float*    Sf  = (float*)(smem + SM_WSL);
  _Float16* Ps  = (_Float16*)(smem + SM_PS);
  _Float16* qs  = (_Float16*)(smem + SM_QS);
  _Float16* ks  = (_Float16*)(smem + SM_KS);
  float*    os  = (float*)(smem + SM_QS);      // 64*132 f32 = 33792 <= 34816
  _Float16* vT  = (_Float16*)(smem + SM_VT);
  float*    bt  = (float*)(smem + SM_BT);

  int win = blockIdx.x;
  int b   = win >> 10;
  int wh  = (win >> 5) & 31;
  int ww  = win & 31;
  int tid  = threadIdx.x;
  int lane = tid & 31;
  int wv   = tid >> 5;  // 8 waves

  __builtin_prefetch(w_qkv, 0, 0);  // global_prefetch_b8 on the weight stream

  // ---- Phase 0a: stage x window (float4 reads) + bias table --------------
  for (int idx4 = tid; idx4 < 64 * 128 / 4; idx4 += 256) {
    int c  = idx4 >> 4;
    int p0 = (idx4 & 15) * 4;
    int y  = wh * 8 + (p0 >> 3);
    int xc = ww * 8 + (p0 & 7);
    v4f v = *(const v4f*)&x[(((size_t)b * CCH + c) * 256 + y) * 256 + xc];
#pragma unroll
    for (int u = 0; u < 4; ++u) xs[(p0 + u) * 136 + c] = (_Float16)v[u];
  }
  for (int i = tid; i < 225 * HEADS; i += 256) bt[i] = bias_table[i];

  // ---- Phase 0b: QKV projection, results to LDS (q,k row-major; v^T) -----
  for (int which = 0; which < 3; ++which) {
    __syncthreads();  // xs ready / previous chunk compute done
    for (int idx4 = tid; idx4 < 128 * 128 / 4; idx4 += 256) {
      int c3 = idx4 >> 5;
      int c0 = (idx4 & 31) * 4;
      v4f v = *(const v4f*)&w_qkv[(size_t)which * 16384 + c3 * 128 + c0];
      v4h hv;
#pragma unroll
      for (int u = 0; u < 4; ++u) hv[u] = (_Float16)v[u];
      *(v4h*)&wsl[c3 * 136 + c0] = hv;
    }
    __syncthreads();

    // 4 M-tiles x 8 N-tiles, 4 per wave, K=128 (4 wmma steps)
#pragma unroll
    for (int i = 0; i < 4; ++i) {
      int t  = wv * 4 + i;
      int mt = t >> 3;  // constant per wave -> A-frags CSE across i
      int nt = t & 7;
      v8f acc = {};
#pragma unroll
      for (int kk = 0; kk < 4; ++kk) {
        v16h a  = frag_ld(xs, mt * 16, kk * 32, 136, lane);
        v16h bb = frag_ld(wsl, nt * 16, kk * 32, 136, lane);
        acc = __builtin_amdgcn_wmma_f32_16x16x32_f16(false, a, false, bb,
                                                     (short)0, acc, false, false);
      }
      int c3l = nt * 16 + (lane & 15);
      int h   = lane >> 4;
      if (which == 2) {
        // D rows are contiguous in p for a lane -> one packed 16B LDS store
        v8h o;
#pragma unroll
        for (int r = 0; r < 8; ++r) o[r] = (_Float16)acc[r];
        *(v8h*)&vT[c3l * 72 + mt * 16 + 8 * h] = o;
      } else {
        _Float16* dst = (which == 0) ? qs : ks;
#pragma unroll
        for (int r = 0; r < 8; ++r)
          dst[(mt * 16 + r + 8 * h) * 136 + c3l] = (_Float16)acc[r];
      }
    }
  }
  __syncthreads();  // (A) QKV done: xs & wsl regions free, q/k/vT valid

  // ---- Phase 1: attention, one head at a time (all 8 waves) --------------
  const float scale = 0.17677669529663687f;  // 1/sqrt(32)
  for (int head = 0; head < HEADS; ++head) {
    // S = q @ k^T : 16 tiles / 8 waves = 2 each (K=32 -> 1 wmma per tile)
    {
      int mt = wv >> 1;
      v16h a = frag_ld(qs, mt * 16, head * HD, 136, lane);
#pragma unroll
      for (int q2 = 0; q2 < 2; ++q2) {
        int nt = (wv & 1) * 2 + q2;
        v16h bb = frag_ld(ks, nt * 16, head * HD, 136, lane);
        v8f acc = {};
        acc = __builtin_amdgcn_wmma_f32_16x16x32_f16(false, a, false, bb,
                                                     (short)0, acc, false, false);
        int n = nt * 16 + (lane & 15);
        int h = lane >> 4;
#pragma unroll
        for (int r = 0; r < 8; ++r) Sf[(mt * 16 + r + 8 * h) * 68 + n] = acc[r];
      }
    }
    __syncthreads();  // Sf ready (also guards Ps reuse ordering across heads)

    // scale + relative-position bias + softmax, register-resident per row
    if (tid < 64) {
      int row = tid;
      float vals[64];
      float mx = -3.4e38f;
#pragma unroll
      for (int g = 0; g < 16; ++g) {
        v4f s4 = *(const v4f*)&Sf[row * 68 + g * 4];
        v4i r4 = *(const v4i*)&rel_index[row * 64 + g * 4];
#pragma unroll
        for (int u = 0; u < 4; ++u) {
          float v = s4[u] * scale + bt[r4[u] * HEADS + head];
          vals[g * 4 + u] = v;
          mx = fmaxf(mx, v);
        }
      }
      float sum = 0.f;
#pragma unroll
      for (int c2 = 0; c2 < 64; ++c2) {
        float e = __expf(vals[c2] - mx);
        vals[c2] = e;
        sum += e;
      }
      float inv = 1.f / sum;
#pragma unroll
      for (int g = 0; g < 16; ++g) {
        v4h p4;
#pragma unroll
        for (int u = 0; u < 4; ++u) p4[u] = (_Float16)(vals[g * 4 + u] * inv);
        *(v4h*)&Ps[row * 72 + g * 4] = p4;
      }
    }
    __syncthreads();  // Ps ready

    // O = P @ V : 4 mt x 2 nt tiles / 8 waves = 1 each, K=64 (2 wmma)
    {
      int mt = wv >> 1;
      int nt = wv & 1;
      v8f acc = {};
#pragma unroll
      for (int kt = 0; kt < 2; ++kt) {
        v16h a  = frag_ld(Ps, mt * 16, kt * 32, 72, lane);
        v16h bb = frag_ld(vT, head * HD + nt * 16, kt * 32, 72, lane);
        acc = __builtin_amdgcn_wmma_f32_16x16x32_f16(false, a, false, bb,
                                                     (short)0, acc, false, false);
      }
      int c = head * HD + nt * 16 + (lane & 15);
      int h = lane >> 4;
#pragma unroll
      for (int r = 0; r < 8; ++r)
        xs[(mt * 16 + r + 8 * h) * 136 + c] = (_Float16)acc[r];  // O into xs region
    }
    // next head's Sf writes are ordered by its own first __syncthreads()
  }
  __syncthreads();  // O complete; Sf/Ps dead -> region becomes w_proj

  // ---- Phase 2: output projection + bias + [B,C,H,W] store ---------------
  for (int idx4 = tid; idx4 < 128 * 128 / 4; idx4 += 256) {
    int co = idx4 >> 5;
    int c0 = (idx4 & 31) * 4;
    v4f v = *(const v4f*)&w_proj[co * 128 + c0];
    v4h hv;
#pragma unroll
    for (int u = 0; u < 4; ++u) hv[u] = (_Float16)v[u];
    *(v4h*)&wsl[co * 136 + c0] = hv;  // natural [c_out][c_in]
  }
  __syncthreads();

  v8f acc[4];
  int mts[4], nts[4];
#pragma unroll
  for (int i = 0; i < 4; ++i) {
    int t  = wv * 4 + i;
    int mt = t >> 3;
    int nt = t & 7;
    mts[i] = mt;
    nts[i] = nt;
    v8f a_ = {};
#pragma unroll
    for (int kk = 0; kk < 4; ++kk) {
      v16h a  = frag_ld(xs, mt * 16, kk * 32, 136, lane);   // O
      v16h bb = frag_ld(wsl, nt * 16, kk * 32, 136, lane);  // B[k][n]=wp[n][k]
      a_ = __builtin_amdgcn_wmma_f32_16x16x32_f16(false, a, false, bb,
                                                  (short)0, a_, false, false);
    }
    acc[i] = a_;
  }
  // os reuses qs/ks region (dead since phase 1); no conflict with xs/wsl reads
#pragma unroll
  for (int i = 0; i < 4; ++i) {
    int n = nts[i] * 16 + (lane & 15);
    int h = lane >> 4;
#pragma unroll
    for (int r = 0; r < 8; ++r) os[(mts[i] * 16 + r + 8 * h) * 132 + n] = acc[i][r];
  }
  __syncthreads();

  // float4 stores: per channel, window rows are 2x float4-contiguous runs.
  for (int idx4 = tid; idx4 < 64 * 128 / 4; idx4 += 256) {
    int c  = idx4 >> 4;
    int p0 = (idx4 & 15) * 4;
    int y  = wh * 8 + (p0 >> 3);
    int xc = ww * 8 + (p0 & 7);
    float bp = b_proj[c];
    v4f o;
#pragma unroll
    for (int u = 0; u < 4; ++u) o[u] = os[(p0 + u) * 132 + c] + bp;
    *(v4f*)&out[(((size_t)b * CCH + c) * 256 + y) * 256 + xc] = o;
  }
}

// ---------------------------------------------------------------------------
extern "C" void kernel_launch(void* const* d_in, const int* in_sizes, int n_in,
                              void* d_out, int out_size, void* d_ws, size_t ws_size,
                              hipStream_t stream) {
  (void)in_sizes; (void)n_in; (void)out_size; (void)d_ws; (void)ws_size;
  const float* x          = (const float*)d_in[0];
  const float* w_qkv      = (const float*)d_in[1];
  const float* bias_table = (const float*)d_in[2];
  const float* w_proj     = (const float*)d_in[3];
  const float* b_proj     = (const float*)d_in[4];
  const int*   rel_index  = (const int*)d_in[5];
  float* out = (float*)d_out;

  fused_window_attn_kernel<<<NWIN, 256, 0, stream>>>(
      x, w_qkv, bias_table, rel_index, w_proj, b_proj, out);
}
